// GAT_75849122447741
// MI455X (gfx1250) — compile-verified, matrix operand order checked by reading, not run on hardware
//
#include <hip/hip_runtime.h>

#define DEVINL __device__ __forceinline__

typedef __attribute__((ext_vector_type(16))) __bf16          v16bf;
typedef __attribute__((ext_vector_type(16))) unsigned short  v16us;
typedef __attribute__((ext_vector_type(8)))  float           v8f;
typedef __attribute__((ext_vector_type(4)))  unsigned int    v4u;
typedef __attribute__((ext_vector_type(8)))  int             v8i;
typedef __attribute__((ext_vector_type(4)))  int             v4i;

// ---------------- problem constants ----------------
constexpr int Nn   = 50000;          // nodes
constexpr int Ee   = 800000;         // edges (without self loops)
constexpr int Aa   = Ee + Nn;        // edges incl. self loops = 850000
constexpr int Dd   = 128;            // input node dim
constexpr int Hh   = 4;              // heads
constexpr int Cc   = 64;             // channels per head
constexpr int HC   = 256;            // H*C
constexpr int Bb   = 64;             // graphs
constexpr int OUTD = 10;

// TDM availability (codegen-verified on ROCm 7.2 / clang-22 5-arg; clang-23 6-arg)
#if defined(__has_builtin)
#if __has_builtin(__builtin_amdgcn_tensor_load_to_lds) && \
    __has_builtin(__builtin_amdgcn_s_wait_tensorcnt)
#define USE_TDM 1
#endif
#endif
#ifndef USE_TDM
#define USE_TDM 0
#endif

// ---------------- helpers ----------------
DEVINL unsigned short f2bf(float f) {       // fp32 -> bf16, round-to-nearest-even
  unsigned int u = __float_as_uint(f);
  u += 0x7FFFu + ((u >> 16) & 1u);
  return (unsigned short)(u >> 16);
}

DEVINL float leakyr(float v) { return v > 0.f ? v : 0.2f * v; }

DEVINL void edge_sd(const int* __restrict__ ei, int e, int& s, int& d) {
  if (e < Ee) { s = ei[e]; d = ei[Ee + e]; }
  else        { s = e - Ee; d = s; }        // self loop
}

DEVINL void atomicMaxF(float* addr, float val) {
  unsigned int* ua = (unsigned int*)addr;
  unsigned int old = *((volatile unsigned int*)ua);
  while (__uint_as_float(old) < val) {
    unsigned int assumed = old;
    old = atomicCAS(ua, assumed, __float_as_uint(val));
    if (old == assumed) break;
  }
}

// ---------------- weight pre-pack: fp32 [K,256] -> bf16 WMMA B-fragments ----------------
// one 32-thread block per fragment (kt, nt); fragment = 32 lanes x 16 bf16, contiguous.
__global__ __launch_bounds__(32)
void gat_prep_w(const float* __restrict__ W, unsigned short* __restrict__ out) {
  const int lane = threadIdx.x;             // 0..31
  const int frag = blockIdx.x;              // kt*16 + nt
  const int kt = frag >> 4, nt = frag & 15;
  const int col = nt * 16 + (lane & 15);
  const int kb  = kt * 32 + ((lane >> 4) ? 8 : 0);
  unsigned short vals[16];
#pragma unroll
  for (int i = 0; i < 8; ++i) vals[i]     = f2bf(W[(size_t)(kb + i)      * HC + col]);
#pragma unroll
  for (int i = 0; i < 8; ++i) vals[8 + i] = f2bf(W[(size_t)(kb + 16 + i) * HC + col]);
  unsigned short* o = out + ((size_t)frag * 32 + lane) * 16;
#pragma unroll
  for (int i = 0; i < 16; ++i) o[i] = vals[i];
}

// ---------------- WMMA GEMM: Hout[N,256] = X[N,K] @ W[K,256] ----------------
// block = 128 threads (4 waves). wave w owns output columns [w*64, w*64+64) (== head w).
// grid.x = N/16 row tiles. The 16-row X tile (contiguous 16*K floats) is staged into LDS
// once per block — via the Tensor Data Mover when available — with a 4-dword row pad so
// that A-fragment ds_load_b128 reads are bank-conflict-free (rows stride K+4 dwords).
template <int K>
__global__ __launch_bounds__(128)
void gat_gemm_wmma(const float* __restrict__ X,
                   const unsigned short* __restrict__ wfrag,
                   float* __restrict__ Hout) {
  static_assert(K == 128 || K == 256, "K must be 128 or 256");
  constexpr int LDSK = K + 4;               // padded LDS row stride (floats)
  __shared__ float xtile[16 * LDSK];

  const int lane  = threadIdx.x & 31;
  const int wave  = threadIdx.x >> 5;
  const int tile  = blockIdx.x;

#if USE_TDM
  if (threadIdx.x < 32) {                   // one wave issues the TDM descriptor
    unsigned long long ga =
        (unsigned long long)(const void*)(X + (size_t)tile * 16 * K);
    unsigned ldsoff = (unsigned)(unsigned long long)(const void*)&xtile[0];
    const unsigned nel = 16u * K;           // 4-byte elements, contiguous
    const unsigned padInterval = (K == 256) ? 7u : 6u;  // pad every 256 / 128 dwords
    const unsigned padAmount   = 3u;                    // insert 4 dwords
    // D# group 0: count=1 | lds_addr | global_addr | type=2
    v4u g0;
    g0[0] = 1u;
    g0[1] = ldsoff;
    g0[2] = (unsigned)(ga & 0xffffffffu);
    g0[3] = (unsigned)((ga >> 32) & 0x01ffffffu) | (2u << 30);
    // D# group 1: data_size=4B, pad_enable, 1-D tensor/tile of nel elements
    v8i g1;
    g1[0] = (int)((2u << 16) | (1u << 20) | (padInterval << 22) | (padAmount << 25));
    g1[1] = (int)((nel & 0xffffu) << 16);            // tensor_dim0[15:0]
    g1[2] = (int)(((nel >> 16) & 0xffffu) | (1u << 16)); // tensor_dim0[31:16] | tensor_dim1=1
    g1[3] = (int)((nel & 0xffffu) << 16);            // tensor_dim1 hi=0 | tile_dim0=nel
    g1[4] = 0;                                       // tile_dim1=0 (unused), tile_dim2=0
    g1[5] = (int)nel;                                // tensor_dim0_stride
    g1[6] = 0;
    g1[7] = 0;
    v4i g2 = {}; v4i g3 = {};
#if defined(__clang_major__) && __clang_major__ >= 23
    v8i g4 = {};
    __builtin_amdgcn_tensor_load_to_lds(g0, g1, g2, g3, g4, 0);
#else
    __builtin_amdgcn_tensor_load_to_lds(g0, g1, g2, g3, 0);
#endif
    __builtin_amdgcn_s_wait_tensorcnt(0);
  }
  __syncthreads();
  asm volatile("" : : "r"(&xtile[0]) : "memory");   // LDS was written by TDM
#else
  {
    const float4* src = (const float4*)(X + (size_t)tile * 16 * K);
    for (int i = threadIdx.x; i < 16 * K / 4; i += 128) {
      int r = i / (K / 4);
      int c = i - r * (K / 4);
      *(float4*)&xtile[r * LDSK + c * 4] = src[i];
    }
    __syncthreads();
  }
#endif

  const int arow  = lane & 15;
  const int khalf = (lane >> 4) << 3;       // 0 or 8

  v8f zero = {};
  v8f acc[4];
#pragma unroll
  for (int i = 0; i < 4; ++i) acc[i] = zero;

#pragma unroll
  for (int kt = 0; kt < K / 32; ++kt) {
    // A fragment from LDS: lanes 0-15 -> K {0..7,16..23}, lanes 16-31 -> {8..15,24..31}
    const float* xp = &xtile[arow * LDSK + kt * 32 + khalf];
    float4 f0 = *(const float4*)(xp + 0);
    float4 f1 = *(const float4*)(xp + 4);
    float4 f2 = *(const float4*)(xp + 16);
    float4 f3 = *(const float4*)(xp + 20);
    v16us au;
    au[0]  = f2bf(f0.x); au[1]  = f2bf(f0.y); au[2]  = f2bf(f0.z); au[3]  = f2bf(f0.w);
    au[4]  = f2bf(f1.x); au[5]  = f2bf(f1.y); au[6]  = f2bf(f1.z); au[7]  = f2bf(f1.w);
    au[8]  = f2bf(f2.x); au[9]  = f2bf(f2.y); au[10] = f2bf(f2.z); au[11] = f2bf(f2.w);
    au[12] = f2bf(f3.x); au[13] = f2bf(f3.y); au[14] = f2bf(f3.z); au[15] = f2bf(f3.w);
    v16bf a = __builtin_bit_cast(v16bf, au);
#pragma unroll
    for (int nt = 0; nt < 4; ++nt) {
      const int frag = kt * 16 + wave * 4 + nt;
      v16us bu = *(const v16us*)(wfrag + ((size_t)frag * 32 + lane) * 16);
      v16bf b  = __builtin_bit_cast(v16bf, bu);
      acc[nt] = __builtin_amdgcn_wmma_f32_16x16x32_bf16(
          /*neg_a=*/false, a, /*neg_b=*/false, b,
          /*c_mod=*/(short)0, acc[nt], /*reuse_a=*/false, /*reuse_b=*/false);
    }
  }
  // C/D layout: lane 0-15 -> rows 0..7 (vgpr r), col=lane ; lane 16-31 -> rows 8..15, col=lane-16
  const int rbase = tile * 16 + ((lane >> 4) << 3);
  const int col0  = wave * 64 + (lane & 15);
#pragma unroll
  for (int nt = 0; nt < 4; ++nt)
#pragma unroll
    for (int r = 0; r < 8; ++r)
      Hout[(size_t)(rbase + r) * HC + col0 + nt * 16] = acc[nt][r];
}

// ---------------- attention coefficients: al_s/al_d [N,H]; also init m/den ----------------
// wave per (node, head); lane covers 2 channels.
__global__ __launch_bounds__(256)
void gat_coeffs(const float* __restrict__ Hf,
                const float* __restrict__ as, const float* __restrict__ ad,
                float* __restrict__ als, float* __restrict__ ald,
                float* __restrict__ mmax, float* __restrict__ den) {
  const int wid  = blockIdx.x * 8 + (threadIdx.x >> 5);
  const int lane = threadIdx.x & 31;
  if (wid >= Nn * Hh) return;
  const int node = wid >> 2, h = wid & 3;
  const float* hp = Hf + (size_t)node * HC + h * Cc;
  float h0 = hp[lane], h1 = hp[lane + 32];
  float s = h0 * as[h * Cc + lane] + h1 * as[h * Cc + lane + 32];
  float d = h0 * ad[h * Cc + lane] + h1 * ad[h * Cc + lane + 32];
#pragma unroll
  for (int off = 16; off > 0; off >>= 1) {
    s += __shfl_xor(s, off, 32);
    d += __shfl_xor(d, off, 32);
  }
  if (lane == 0) {
    als[wid] = s; ald[wid] = d;
    mmax[wid] = -1e30f; den[wid] = 0.f;
  }
}

// ---------------- edge softmax: max, then exp-sum ----------------
__global__ __launch_bounds__(256)
void gat_edge_max(const int* __restrict__ ei,
                  const float* __restrict__ als, const float* __restrict__ ald,
                  float* __restrict__ mmax) {
  const int tid = blockIdx.x * 256 + threadIdx.x;
  if (tid >= Aa * Hh) return;
  const int e = tid >> 2, h = tid & 3;
  int s, d; edge_sd(ei, e, s, d);
  float v = leakyr(als[s * Hh + h] + ald[d * Hh + h]);
  atomicMaxF(&mmax[d * Hh + h], v);
}

__global__ __launch_bounds__(256)
void gat_edge_den(const int* __restrict__ ei,
                  const float* __restrict__ als, const float* __restrict__ ald,
                  const float* __restrict__ mmax, float* __restrict__ den) {
  const int tid = blockIdx.x * 256 + threadIdx.x;
  if (tid >= Aa * Hh) return;
  const int e = tid >> 2, h = tid & 3;
  int s, d; edge_sd(ei, e, s, d);
  const int idx = d * Hh + h;
  float v = leakyr(als[s * Hh + h] + ald[idx]);
  atomicAdd(&den[idx], __expf(v - mmax[idx]));
}

// ---------------- message aggregation: wave per (edge, head), 2 channels/lane ----------------
__global__ __launch_bounds__(256)
void gat_aggregate(const int* __restrict__ ei, const float* __restrict__ Hf,
                   const float* __restrict__ als, const float* __restrict__ ald,
                   const float* __restrict__ mmax, const float* __restrict__ den,
                   float* __restrict__ Agg) {
  const int wid  = blockIdx.x * 8 + (threadIdx.x >> 5);
  const int lane = threadIdx.x & 31;
  if (wid >= Aa * Hh) return;
  const int e = wid >> 2, h = wid & 3;
  int s, d; edge_sd(ei, e, s, d);
  const int idx = d * Hh + h;
  float v = leakyr(als[s * Hh + h] + ald[idx]);
  float w = __expf(v - mmax[idx]) / (den[idx] + 1e-16f);
  const float* hs = Hf + (size_t)s * HC + h * Cc;
  float* ag = Agg + (size_t)d * HC + h * Cc;
  atomicAdd(&ag[lane],      hs[lane]      * w);
  atomicAdd(&ag[lane + 32], hs[lane + 32] * w);
}

// ---------------- epilogues ----------------
__global__ __launch_bounds__(256)
void gat_fill(float* __restrict__ p, float v, long n) {
  long stride = (long)gridDim.x * blockDim.x;
  for (long i = (long)blockIdx.x * blockDim.x + threadIdx.x; i < n; i += stride) p[i] = v;
}

__global__ __launch_bounds__(256)
void gat_bias_elu(const float* __restrict__ Agg, const float* __restrict__ bias,
                  float* __restrict__ Xout) {
  const long i = (long)blockIdx.x * 256 + threadIdx.x;   // N*HC total, exact
  float v = Agg[i] + bias[i & (HC - 1)];
  Xout[i] = v > 0.f ? v : (__expf(v) - 1.f);
}

__global__ __launch_bounds__(256)
void gat_mean_heads(const float* __restrict__ Agg, const float* __restrict__ bias,
                    float* __restrict__ Hfin) {
  const long i = (long)blockIdx.x * 256 + threadIdx.x;   // N*C total, exact
  const long n = i >> 6; const int c = (int)(i & 63);
  const float* a = Agg + n * HC;
  Hfin[i] = 0.25f * (a[c] + a[Cc + c] + a[2 * Cc + c] + a[3 * Cc + c]) + bias[c];
}

// ---------------- global mean pool ----------------
__global__ __launch_bounds__(256)
void gat_pool(const float* __restrict__ Hfin, const int* __restrict__ batch,
              float* __restrict__ pooled, float* __restrict__ cnt) {
  const int node = blockIdx.x * 8 + (threadIdx.x >> 5);
  const int lane = threadIdx.x & 31;
  if (node >= Nn) return;
  const int b = batch[node];
  atomicAdd(&pooled[b * Cc + lane],      Hfin[(size_t)node * Cc + lane]);
  atomicAdd(&pooled[b * Cc + lane + 32], Hfin[(size_t)node * Cc + lane + 32]);
  if (lane == 0) atomicAdd(&cnt[b], 1.0f);
}

// ---------------- predictor MLP, one block ----------------
__global__ __launch_bounds__(64)
void gat_mlp(const float* __restrict__ pooled, const float* __restrict__ cnt,
             const float* __restrict__ pW1, const float* __restrict__ pb1,
             const float* __restrict__ pW2, const float* __restrict__ pb2,
             float* __restrict__ out) {
  const int b = threadIdx.x;
  float cb = cnt[b]; cb = cb > 1.f ? cb : 1.f;
  float p[Cc];
#pragma unroll
  for (int c = 0; c < Cc; ++c) p[c] = pooled[b * Cc + c] / cb;
  float z[32];
  for (int j = 0; j < 32; ++j) {
    float acc = pb1[j];
    for (int c = 0; c < Cc; ++c) acc += p[c] * pW1[c * 32 + j];
    z[j] = acc > 0.f ? acc : 0.f;
  }
  for (int o = 0; o < OUTD; ++o) {
    float acc = pb2[o];
    for (int j = 0; j < 32; ++j) acc += z[j] * pW2[j * OUTD + o];
    out[b * OUTD + o] = acc;
  }
}

// ---------------- host orchestration ----------------
extern "C" void kernel_launch(void* const* d_in, const int* in_sizes, int n_in,
                              void* d_out, int out_size, void* d_ws, size_t ws_size,
                              hipStream_t stream) {
  (void)in_sizes; (void)n_in; (void)out_size; (void)ws_size;
  const float* x   = (const float*)d_in[0];
  const int*   ei  = (const int*)  d_in[1];
  const int*   bat = (const int*)  d_in[2];
  const float* W0  = (const float*)d_in[3];
  const float* b0  = (const float*)d_in[4];
  const float* as0 = (const float*)d_in[5];
  const float* ad0 = (const float*)d_in[6];
  const float* W1  = (const float*)d_in[7];
  const float* b1  = (const float*)d_in[8];
  const float* as1 = (const float*)d_in[9];
  const float* ad1 = (const float*)d_in[10];
  const float* W2  = (const float*)d_in[11];
  const float* b2  = (const float*)d_in[12];
  const float* as2 = (const float*)d_in[13];
  const float* ad2 = (const float*)d_in[14];
  const float* pW1 = (const float*)d_in[15];
  const float* pb1 = (const float*)d_in[16];
  const float* pW2 = (const float*)d_in[17];
  const float* pb2 = (const float*)d_in[18];

  // workspace carve-up (~106 MB)
  char* ws = (char*)d_ws;
  unsigned short* wfrag = (unsigned short*)ws;            // 327,680 B of packed weights
  float* bufA   = (float*)(ws + 0x60000);                 // [N,256] h (pre-aggregation)
  float* bufB   = bufA + (size_t)Nn * HC;                 // [N,256] x-in / agg, ping-pong
  float* als    = bufB + (size_t)Nn * HC;                 // [N,H]
  float* ald    = als + (size_t)Nn * Hh;
  float* mmax   = ald + (size_t)Nn * Hh;
  float* den    = mmax + (size_t)Nn * Hh;
  float* pooled = den + (size_t)Nn * Hh;                  // [B,64]
  float* cnt    = pooled + Bb * Cc;                       // [B]

  unsigned short* wf0 = wfrag;                            // 64 frags
  unsigned short* wf1 = wfrag + 32768;                    // 128 frags
  unsigned short* wf2 = wfrag + 98304;                    // 128 frags

  const int EH_BLOCKS  = (Aa * Hh + 255) / 256;           // thread-per-(edge,head)
  const int EHW_BLOCKS = (Aa * Hh) / 8;                   // wave-per-(edge,head), exact

  gat_prep_w<<<(Dd / 32) * 16, 32, 0, stream>>>(W0, wf0);
  gat_prep_w<<<(HC / 32) * 16, 32, 0, stream>>>(W1, wf1);
  gat_prep_w<<<(HC / 32) * 16, 32, 0, stream>>>(W2, wf2);

  // ---- layer 0 (K = 128, concat + ELU) ----
  gat_gemm_wmma<Dd><<<Nn / 16, 128, 0, stream>>>(x, wf0, bufA);
  gat_coeffs<<<(Nn * Hh) / 8, 256, 0, stream>>>(bufA, as0, ad0, als, ald, mmax, den);
  gat_edge_max<<<EH_BLOCKS, 256, 0, stream>>>(ei, als, ald, mmax);
  gat_edge_den<<<EH_BLOCKS, 256, 0, stream>>>(ei, als, ald, mmax, den);
  gat_fill<<<2048, 256, 0, stream>>>(bufB, 0.f, (long)Nn * HC);
  gat_aggregate<<<EHW_BLOCKS, 256, 0, stream>>>(ei, bufA, als, ald, mmax, den, bufB);
  gat_bias_elu<<<(Nn * HC) / 256, 256, 0, stream>>>(bufB, b0, bufB);

  // ---- layer 1 (K = 256, concat + ELU) ----
  gat_gemm_wmma<HC><<<Nn / 16, 128, 0, stream>>>(bufB, wf1, bufA);
  gat_coeffs<<<(Nn * Hh) / 8, 256, 0, stream>>>(bufA, as1, ad1, als, ald, mmax, den);
  gat_edge_max<<<EH_BLOCKS, 256, 0, stream>>>(ei, als, ald, mmax);
  gat_edge_den<<<EH_BLOCKS, 256, 0, stream>>>(ei, als, ald, mmax, den);
  gat_fill<<<2048, 256, 0, stream>>>(bufB, 0.f, (long)Nn * HC);
  gat_aggregate<<<EHW_BLOCKS, 256, 0, stream>>>(ei, bufA, als, ald, mmax, den, bufB);
  gat_bias_elu<<<(Nn * HC) / 256, 256, 0, stream>>>(bufB, b1, bufB);

  // ---- layer 2 (K = 256, head-mean, no activation) ----
  gat_gemm_wmma<HC><<<Nn / 16, 128, 0, stream>>>(bufB, wf2, bufA);
  gat_coeffs<<<(Nn * Hh) / 8, 256, 0, stream>>>(bufA, as2, ad2, als, ald, mmax, den);
  gat_edge_max<<<EH_BLOCKS, 256, 0, stream>>>(ei, als, ald, mmax);
  gat_edge_den<<<EH_BLOCKS, 256, 0, stream>>>(ei, als, ald, mmax, den);
  gat_fill<<<2048, 256, 0, stream>>>(bufB, 0.f, (long)Nn * HC);
  gat_aggregate<<<EHW_BLOCKS, 256, 0, stream>>>(ei, bufA, als, ald, mmax, den, bufB);
  gat_mean_heads<<<(Nn * Cc) / 256, 256, 0, stream>>>(bufB, b2, bufA);  // bufA := h_final [N,64]

  // ---- global mean pool + MLP ----
  gat_fill<<<17, 256, 0, stream>>>(pooled, 0.f, (long)(Bb * Cc + Bb));
  gat_pool<<<Nn / 8, 256, 0, stream>>>(bufA, bat, pooled, cnt);
  gat_mlp<<<1, Bb, 0, stream>>>(pooled, cnt, pW1, pb1, pW2, pb2, (float*)d_out);
}
